// Encoder_60052232732804
// MI455X (gfx1250) — matrix operand order repaired and checked by reading, not compiled, not run on hardware
//
#include <hip/hip_runtime.h>
#include <math.h>

typedef __bf16 bf16_t;
typedef __attribute__((ext_vector_type(16))) __bf16 v16bf;
typedef __attribute__((ext_vector_type(8)))  __bf16 v8bf;
typedef __attribute__((ext_vector_type(8)))  float  v8f;
typedef __attribute__((ext_vector_type(4)))  unsigned int u32x4;
typedef __attribute__((ext_vector_type(8)))  int i32x8;
typedef __attribute__((ext_vector_type(4)))  int i32x4;

#define B_ 256
#define T_ 256
#define E_ 512
#define Z_ 256
#define H_ 1024

// ---- TDM availability (probe-confirmed builtins; arity differs per toolchain) ----
#ifndef ENABLE_TDM
#define ENABLE_TDM 1
#endif
#if ENABLE_TDM && defined(__has_builtin)
# if __has_builtin(__builtin_amdgcn_tensor_load_to_lds) && __has_builtin(__builtin_amdgcn_s_wait_tensorcnt)
#  define HAVE_TDM 1
# endif
#endif
#ifndef HAVE_TDM
# define HAVE_TDM 0
#endif
#if HAVE_TDM
# if __has_include(<hip/amd_detail/amd_gfx1250_TDM.h>)
#  define TDM_ARGS6 1
# else
#  define TDM_ARGS6 0
# endif
#endif

union V16U { v16bf v; v8bf h[2]; };

#define LDSROW 40  // 32 K elems + 8 pad (80B rows, 16B aligned, breaks bank conflicts)

// ---------------- fp32 -> bf16 conversion ----------------
__global__ void cvt_f32_bf16(const float* __restrict__ src, bf16_t* __restrict__ dst, int n) {
    int i = blockIdx.x * blockDim.x + threadIdx.x;
    int stride = gridDim.x * blockDim.x;
    for (; i < n; i += stride) dst[i] = (bf16_t)src[i];
}

__global__ void init_state(const float* __restrict__ h0, const float* __restrict__ c0,
                           bf16_t* __restrict__ hbf, float* __restrict__ c, int n) {
    int i = blockIdx.x * blockDim.x + threadIdx.x;
    if (i < n) { hbf[i] = (bf16_t)h0[i]; c[i] = c0[i]; }
}

__global__ void bias_sum(const float* __restrict__ a, const float* __restrict__ b,
                         float* __restrict__ o, int n) {
    int i = blockIdx.x * blockDim.x + threadIdx.x;
    if (i < n) o[i] = a[i] + b[i];
}

// ---------------- helpers ----------------
__device__ __forceinline__ const bf16_t* wtile_ptr(int k, int K1,
        const bf16_t* W1, int ldw1, const bf16_t* W2, int ldw2,
        int n0, int& ldw) {
    if (k < K1) { ldw = ldw1; return W1 + (size_t)n0 * ldw1 + k; }
    ldw = ldw2;  return W2 + (size_t)n0 * ldw2 + (k - K1);
}

#if HAVE_TDM
// DMA a 64-row x 32-elem bf16 tile (row stride ldw elems) into LDS at byte addr
// lds_off, writing LDSROW(=40)-elem padded rows: pad 4 DWORDs after every 16 DWORDs.
__device__ __forceinline__ void tdm_load_tile(const bf16_t* gsrc, int ldw, unsigned lds_off) {
    unsigned long long ga = (unsigned long long)(uintptr_t)gsrc;
    u32x4 g0;
    g0[0] = 1u;                                   // count=1 (valid user descriptor)
    g0[1] = lds_off;                              // lds_addr (bytes)
    g0[2] = (unsigned)ga;                         // global_addr[31:0]
    g0[3] = (unsigned)(ga >> 32) | 0x80000000u;   // global_addr[56:32] | type=2
    i32x8 g1;
    // data_size=1 (2B) | pad_enable | pad_interval=3 (16 DW) | pad_amount=3 (4 DW)
    g1[0] = (int)((1u << 16) | (1u << 20) | (3u << 22) | (3u << 25));
    unsigned td0 = (unsigned)ldw;                 // tensor_dim0 (elems/row)
    unsigned td1 = 8192u;                         // tensor_dim1 (rows upper bound)
    g1[1] = (int)((td0 & 0xFFFFu) << 16);         // [47:32]=atomic_addr=0, [63:48]=td0.lo
    g1[2] = (int)((td0 >> 16) | ((td1 & 0xFFFFu) << 16));
    g1[3] = (int)((td1 >> 16) | (32u << 16));     // tile_dim0 = 32
    g1[4] = (int)(64u);                           // tile_dim1 = 64, tile_dim2 = 0
    g1[5] = (int)ldw;                             // tensor_dim0_stride[31:0]
    g1[6] = 0;                                    // stride[47:32]=0, dim1_stride.lo=0
    g1[7] = 0;
    i32x4 z4 = {0, 0, 0, 0};
#if TDM_ARGS6
    i32x8 z8 = {0, 0, 0, 0, 0, 0, 0, 0};
    __builtin_amdgcn_tensor_load_to_lds(g0, g1, z4, z4, z8, 0);
#else
    __builtin_amdgcn_tensor_load_to_lds(g0, g1, z4, z4, 0);
#endif
}
#endif

// ---------------- fused per-step gates GEMM ----------------
// blockIdx.x in [0,64):  g4[m,n] = sum_k [x_t|h][m,k] * [Wi|Wh][n,k] + bias4[n]
// blockIdx.x in [64,96): g2[m,n] = sum_k  x_z[m,k]  *    Wz[n,k]    + bz[n]
// 256 threads = 8 waves; block tile 128M x 64N; wave tile 16M x 64N; double-buffered LDS W.
__global__ void __launch_bounds__(256)
step_gemm(const bf16_t* __restrict__ Axd, const bf16_t* __restrict__ Axz,
          const bf16_t* __restrict__ hbf,
          const bf16_t* __restrict__ Wi, const bf16_t* __restrict__ Wh,
          const bf16_t* __restrict__ Wz,
          const float* __restrict__ bias4, const float* __restrict__ bz,
          float* __restrict__ g4, float* __restrict__ g2)
{
    __shared__ __align__(16) bf16_t ldsW[2][64 * LDSROW];

    const bf16_t* A1; int lda1, K1;
    const bf16_t* A2; int lda2, K2;
    const bf16_t* W1; int ldw1; const bf16_t* W2; int ldw2;
    const float* bias; float* out; int ldo; int n0;
    if (blockIdx.x < 64) {
        A1 = Axd; lda1 = T_ * E_; K1 = E_;
        A2 = hbf; lda2 = H_;      K2 = H_;
        W1 = Wi;  ldw1 = E_;      W2 = Wh; ldw2 = H_;
        bias = bias4; out = g4; ldo = 4 * H_;
        n0 = blockIdx.x * 64;
    } else {
        A1 = Axz; lda1 = T_ * Z_; K1 = Z_;
        A2 = nullptr; lda2 = 0;   K2 = 0;
        W1 = Wz;  ldw1 = Z_;      W2 = nullptr; ldw2 = 0;
        bias = bz; out = g2; ldo = 2 * H_;
        n0 = (blockIdx.x - 64) * 64;
    }

    const int tid   = threadIdx.x;
    const int lane  = tid & 31;
    const int wv    = tid >> 5;
    const int m0    = blockIdx.y * 128 + wv * 16;
    const int khalf = lane >> 4;
    const int mrow  = m0 + (lane & 15);
    const int Ktot  = K1 + K2;

    v8f acc[4] = {};

    // ---- prologue: stage tile k=0 into buffer 0 ----
#if HAVE_TDM
    if (wv == 0) {
        int ldw; const bf16_t* src = wtile_ptr(0, K1, W1, ldw1, W2, ldw2, n0, ldw);
        tdm_load_tile(src, ldw, (unsigned)(uintptr_t)&ldsW[0][0]);
        __builtin_amdgcn_s_wait_tensorcnt(0);
    }
    __syncthreads();
#else
    {
        int ldw; const bf16_t* wt = wtile_ptr(0, K1, W1, ldw1, W2, ldw2, n0, ldw);
        const int r = tid >> 2, c8 = (tid & 3) * 8;
        v8bf w = *(const v8bf*)(wt + (size_t)r * ldw + c8);
        *(v8bf*)(&ldsW[0][r * LDSROW + c8]) = w;
        __syncthreads();
    }
#endif

    int cur = 0;
    for (int k = 0; k < Ktot; k += 32) {
        const int nxt = cur ^ 1;
        const bool have_next = (k + 32) < Ktot;

        // ---- begin staging next W tile into the other buffer ----
#if HAVE_TDM
        if (have_next && wv == 0) {
            int ldw; const bf16_t* src = wtile_ptr(k + 32, K1, W1, ldw1, W2, ldw2, n0, ldw);
            tdm_load_tile(src, ldw, (unsigned)(uintptr_t)&ldsW[nxt][0]);
        }
#else
        v8bf wreg = {};
        const int r = tid >> 2, c8 = (tid & 3) * 8;
        if (have_next) {
            int ldw; const bf16_t* wt = wtile_ptr(k + 32, K1, W1, ldw1, W2, ldw2, n0, ldw);
            wreg = *(const v8bf*)(wt + (size_t)r * ldw + c8);
        }
#endif

        // ---- A fragment (ISA 16-bit A 16x32 layout) ----
        V16U a;
        {
            const bf16_t* ab; int lda; int kk;
            if (k < K1) { ab = A1; lda = lda1; kk = k; }
            else        { ab = A2; lda = lda2; kk = k - K1; }
            const bf16_t* ap = ab + (size_t)mrow * lda + kk + khalf * 8;
            a.h[0] = *(const v8bf*)(ap);
            a.h[1] = *(const v8bf*)(ap + 16);
        }

        // ---- preload all 4 B fragments, then 4 WMMAs (pipelined dscnt waits) ----
        const int kh2 = khalf * 8;
        V16U bf4[4];
        #pragma unroll
        for (int nt = 0; nt < 4; ++nt) {
            const int col = nt * 16 + (lane & 15);
            bf4[nt].h[0] = *(const v8bf*)(&ldsW[cur][col * LDSROW + kh2]);
            bf4[nt].h[1] = *(const v8bf*)(&ldsW[cur][col * LDSROW + kh2 + 16]);
        }
        #pragma unroll
        for (int nt = 0; nt < 4; ++nt)
            acc[nt] = __builtin_amdgcn_wmma_f32_16x16x32_bf16(
                false, a.v, false, bf4[nt].v, (short)0, acc[nt], false, false);

        // ---- finish staging + one barrier per iteration ----
#if HAVE_TDM
        if (have_next) {
            if (wv == 0) __builtin_amdgcn_s_wait_tensorcnt(0);
            __syncthreads();
        }
#else
        if (have_next) {
            *(v8bf*)(&ldsW[nxt][r * LDSROW + c8]) = wreg;
            __syncthreads();
        }
#endif
        cur = nxt;
    }

    // ---- epilogue: D layout VGPR i -> row m0 + i + 8*khalf ----
    #pragma unroll
    for (int nt = 0; nt < 4; ++nt) {
        const int col = n0 + nt * 16 + (lane & 15);
        const float b = bias[col];
        #pragma unroll
        for (int i = 0; i < 8; ++i)
            out[(size_t)(m0 + i + 8 * khalf) * ldo + col] = acc[nt][i] + b;
    }
}

// ---------------- per-step cell update ----------------
__global__ void cell_update(const float* __restrict__ g4, const float* __restrict__ g2,
                            float* __restrict__ c, bf16_t* __restrict__ hbf,
                            float* __restrict__ hs, float* __restrict__ cs,
                            float* __restrict__ hfin, float* __restrict__ cfin,
                            int t, int last)
{
    int i = blockIdx.x * blockDim.x + threadIdx.x;  // 0 .. B*H-1
    int b = i >> 10;        // / H
    int j = i & (H_ - 1);   // % H

    const float* gb = g4 + (size_t)b * 4 * H_;
    float gi = gb[j];
    float gf = gb[H_ + j];
    float gg = gb[2 * H_ + j];
    float go = gb[3 * H_ + j];
    const float* zb = g2 + (size_t)b * 2 * H_;
    float gl = zb[j];
    float gz = zb[H_ + j];

    float si = 1.f / (1.f + __expf(-gi));
    float sf = 1.f / (1.f + __expf(-gf));
    float so = 1.f / (1.f + __expf(-go));
    float sl = 1.f / (1.f + __expf(-gl));

    float cn = sf * c[i] + si * tanhf(gg) + sl * tanhf(gz);
    float hn = so * tanhf(cn);

    c[i]   = cn;
    hbf[i] = (bf16_t)hn;

    size_t o = ((size_t)b * T_ + t) * H_ + j;
    hs[o] = hn;
    cs[o] = cn;
    if (last) { hfin[i] = hn; cfin[i] = cn; }
}

// ---------------- orchestration ----------------
extern "C" void kernel_launch(void* const* d_in, const int* in_sizes, int n_in,
                              void* d_out, int out_size, void* d_ws, size_t ws_size,
                              hipStream_t stream)
{
    (void)in_sizes; (void)n_in; (void)out_size; (void)ws_size;

    const float* in_d = (const float*)d_in[0];
    const float* in_z = (const float*)d_in[1];
    const float* h0   = (const float*)d_in[2];
    const float* c0   = (const float*)d_in[3];
    const float* Wi   = (const float*)d_in[4];
    const float* bi   = (const float*)d_in[5];
    const float* Wh   = (const float*)d_in[6];
    const float* bh   = (const float*)d_in[7];
    const float* Wz   = (const float*)d_in[8];
    const float* bz   = (const float*)d_in[9];

    char* p = (char*)d_ws;
    auto carve = [&](size_t bytes) -> char* {
        char* r = p; p += (bytes + 255) & ~(size_t)255; return r;
    };
    bf16_t* bf_d   = (bf16_t*)carve((size_t)B_ * T_ * E_ * 2);
    bf16_t* bf_z   = (bf16_t*)carve((size_t)B_ * T_ * Z_ * 2);
    bf16_t* bf_Wi  = (bf16_t*)carve((size_t)4 * H_ * E_ * 2);
    bf16_t* bf_Wh  = (bf16_t*)carve((size_t)4 * H_ * H_ * 2);
    bf16_t* bf_Wz  = (bf16_t*)carve((size_t)2 * H_ * Z_ * 2);
    float*  g4     = (float*) carve((size_t)B_ * 4 * H_ * 4);
    float*  g2     = (float*) carve((size_t)B_ * 2 * H_ * 4);
    bf16_t* hbf    = (bf16_t*)carve((size_t)B_ * H_ * 2);
    float*  cst    = (float*) carve((size_t)B_ * H_ * 4);
    float*  bias4  = (float*) carve((size_t)4 * H_ * 4);

    // one-time: bf16 copies, fused bias, state init
    cvt_f32_bf16<<<2048, 256, 0, stream>>>(in_d, bf_d, B_ * T_ * E_);
    cvt_f32_bf16<<<2048, 256, 0, stream>>>(in_z, bf_z, B_ * T_ * Z_);
    cvt_f32_bf16<<<1024, 256, 0, stream>>>(Wi, bf_Wi, 4 * H_ * E_);
    cvt_f32_bf16<<<1024, 256, 0, stream>>>(Wh, bf_Wh, 4 * H_ * H_);
    cvt_f32_bf16<<<512,  256, 0, stream>>>(Wz, bf_Wz, 2 * H_ * Z_);
    bias_sum<<<(4 * H_) / 256, 256, 0, stream>>>(bi, bh, bias4, 4 * H_);
    init_state<<<(B_ * H_) / 256, 256, 0, stream>>>(h0, c0, hbf, cst, B_ * H_);

    float* hs = (float*)d_out;
    float* cs = hs + (size_t)B_ * T_ * H_;
    float* hf = cs + (size_t)B_ * T_ * H_;
    float* cf = hf + (size_t)B_ * H_;

    for (int t = 0; t < T_; ++t) {
        step_gemm<<<dim3(96, 2), 256, 0, stream>>>(
            bf_d + (size_t)t * E_, bf_z + (size_t)t * Z_, hbf,
            bf_Wi, bf_Wh, bf_Wz, bias4, bz, g4, g2);
        cell_update<<<(B_ * H_) / 256, 256, 0, stream>>>(
            g4, g2, cst, hbf, hs, cs, hf, cf, t, t == T_ - 1);
    }
}
